// MyConv2d_18391049961801
// MI455X (gfx1250) — matrix operand order repaired and checked by reading, not compiled, not run on hardware
//
#include <hip/hip_runtime.h>
#include <hip/hip_bf16.h>

typedef __attribute__((ext_vector_type(16))) __bf16 v16bf;
typedef __attribute__((ext_vector_type(8)))  float  v8f;

#define BATCH 16
#define CIN   32
#define HWDIM 256
#define COUT  32
#define HT    4            // output rows per workgroup
#define WT    64           // output cols per workgroup
#define PR    (HT + 2)     // patch rows (halo)
#define PC    (WT + 2)     // patch cols (halo)
#define CSTR  40           // padded c_in stride in shorts (80 B, 16B-aligned, bank-spread)

#define W_ELEMS (9 * COUT * CSTR)      // 11520 shorts per hi/lo weight plane
#define P_ELEMS (PR * PC * CSTR)       // 15840 shorts per hi/lo patch plane

union V16 { v16bf v; uint4 q[2]; };

// Load a 16-element bf16 fragment as two 16-byte LDS reads.
__device__ __forceinline__ v16bf ld_frag(const unsigned short* p, int second_off_shorts) {
    V16 u;
    u.q[0] = *(const uint4*)(p);
    u.q[1] = *(const uint4*)(p + second_off_shorts);
    return u.v;
}

__device__ __forceinline__ void split_bf16(float v, unsigned short& hb, unsigned short& lb) {
    unsigned u = __float_as_uint(v);
    hb = (unsigned short)(u >> 16);                       // truncate -> hi
    float hf = __uint_as_float((unsigned)hb << 16);
    lb = (unsigned short)(__float_as_uint(v - hf) >> 16); // exact residual, truncated
}

__global__ __launch_bounds__(256, 1)
void conv3x3_wmma_bf16x3(const float* __restrict__ x,
                         const float* __restrict__ wgt,
                         const float* __restrict__ bias,
                         float* __restrict__ out) {
    extern __shared__ char smem[];
    unsigned short* Wh = (unsigned short*)smem;     // [9][COUT][CSTR]
    unsigned short* Wl = Wh + W_ELEMS;
    unsigned short* Ph = Wl + W_ELEMS;              // [PR][PC][CSTR]
    unsigned short* Pl = Ph + P_ELEMS;
    float*          bs = (float*)(Pl + P_ELEMS);    // [32]

    const int tid  = threadIdx.x;
    const int lane = tid & 31;
    const int wv   = tid >> 5;

    const int wg  = blockIdx.x;
    const int b   = wg >> 8;                 // 256 tiles per image
    const int rem = wg & 255;
    const int h0  = (rem >> 2) << 2;         // 64 row-tiles of 4
    const int w0  = (rem & 3) << 6;          // 4 col-tiles of 64

    // ---- Stage weights: OIHW -> LDS [tap][cout][cin] as bf16 hi/lo ----
    for (int p = tid; p < COUT * CIN; p += 256) {
        const int co = p >> 5, ci = p & 31;
        const float* src = wgt + p * 9;
        #pragma unroll
        for (int t = 0; t < 9; ++t) {
            unsigned short hb, lb;
            split_bf16(src[t], hb, lb);
            const int idx = (t * COUT + co) * CSTR + ci;
            Wh[idx] = hb; Wl[idx] = lb;
        }
    }
    if (tid < 32) bs[tid] = bias[tid];

    // ---- Stage input halo patch: [row][col][cin], coalesced along W ----
    for (int job = wv; job < CIN * PR; job += 8) {
        const int ci = job / PR;
        const int r  = job % PR;
        const int ir = h0 - 1 + r;
        const bool rok = (unsigned)ir < HWDIM;
        const float* srcrow = x + (((size_t)(b * CIN + ci) * HWDIM + ir) * HWDIM);
        #pragma unroll
        for (int cc = 0; cc < 3; ++cc) {
            const int c = lane + cc * 32;
            if (c < PC) {
                const int icol = w0 - 1 + c;
                float v = 0.0f;
                if (rok && (unsigned)icol < HWDIM) v = srcrow[icol];
                unsigned short hb, lb;
                split_bf16(v, hb, lb);
                const int idx = (r * PC + c) * CSTR + ci;
                Ph[idx] = hb; Pl[idx] = lb;
            }
        }
    }
    __syncthreads();

    // ---- Main: each wave does 4 tile-jobs of 16(cout) x 16(pixels) ----
    const int m   = lane & 15;     // A: M row / B,D: N column
    const int hl  = lane >> 4;     // lane half selects K sub-block
    for (int j = wv; j < 32; j += 8) {
        const int chalf = j & 1;         // couts 0-15 or 16-31 (constant per wave)
        const int pt    = j >> 1;
        const int row   = pt & 3;        // output row in tile
        const int ctile = pt >> 2;       // 16-col subtile 0..3

        // Two accumulators, rotated per tap so consecutive WMMAs are
        // independent (breaks the D->C serial chain; hides the 4-slot
        // bf16 WMMA hazard window without v_nops).
        v8f acc0 = {};
        v8f acc1 = {};
        #pragma unroll
        for (int t = 0; t < 9; ++t) {
            const int kh = t / 3, kw = t % 3;
            // A fragment (16x32 bf16): lanes<16 K=0..15, lanes>=16 K=16..31
            const int aidx = (t * COUT + chalf * 16 + m) * CSTR + hl * 16;
            v16bf a_hi = ld_frag(&Wh[aidx], 8);
            v16bf a_lo = ld_frag(&Wl[aidx], 8);
            // B fragment (32x16 bf16): lanes<16 K=0..15, lanes>=16 K=16..31
            const int r = row + kh;
            const int c = ctile * 16 + kw + m;
            const int bidx = (r * PC + c) * CSTR + hl * 16;
            v16bf b_hi = ld_frag(&Ph[bidx], 8);
            v16bf b_lo = ld_frag(&Pl[bidx], 8);

            if ((t & 1) == 0) {
                acc0 = __builtin_amdgcn_wmma_f32_16x16x32_bf16(
                           false, a_hi, false, b_hi, (short)0, acc0, false, false);
                acc1 = __builtin_amdgcn_wmma_f32_16x16x32_bf16(
                           false, a_hi, false, b_lo, (short)0, acc1, false, false);
                acc0 = __builtin_amdgcn_wmma_f32_16x16x32_bf16(
                           false, a_lo, false, b_hi, (short)0, acc0, false, false);
            } else {
                acc1 = __builtin_amdgcn_wmma_f32_16x16x32_bf16(
                           false, a_hi, false, b_hi, (short)0, acc1, false, false);
                acc0 = __builtin_amdgcn_wmma_f32_16x16x32_bf16(
                           false, a_hi, false, b_lo, (short)0, acc0, false, false);
                acc1 = __builtin_amdgcn_wmma_f32_16x16x32_bf16(
                           false, a_lo, false, b_hi, (short)0, acc1, false, false);
            }
        }

        // ---- Epilogue: C/D layout -> NCHW store + bias ----
        const int colo = w0 + ctile * 16 + m;
        const int rowo = h0 + row;
        #pragma unroll
        for (int v = 0; v < 8; ++v) {
            const int co = chalf * 16 + v + hl * 8;
            out[((size_t)(b * COUT + co) * HWDIM + rowo) * HWDIM + colo] =
                acc0[v] + acc1[v] + bs[co];
        }
    }
}

extern "C" void kernel_launch(void* const* d_in, const int* in_sizes, int n_in,
                              void* d_out, int out_size, void* d_ws, size_t ws_size,
                              hipStream_t stream) {
    const float* x    = (const float*)d_in[0];
    const float* wgt  = (const float*)d_in[1];
    const float* bias = (const float*)d_in[2];
    float* out = (float*)d_out;

    const size_t shmem = (size_t)(2 * W_ELEMS + 2 * P_ELEMS) * sizeof(unsigned short)
                       + 32 * sizeof(float); // ~109.6 KB, fits 320 KB WGP LDS
    hipFuncSetAttribute((const void*)conv3x3_wmma_bf16x3,
                        hipFuncAttributeMaxDynamicSharedMemorySize, (int)shmem);

    const int grid = BATCH * (HWDIM / HT) * (HWDIM / WT); // 16 * 64 * 4 = 4096
    conv3x3_wmma_bf16x3<<<grid, 256, shmem, stream>>>(x, wgt, bias, out);
}
